// CFM_8186207666340
// MI455X (gfx1250) — matrix operand order repaired and checked by reading, not compiled, not run on hardware
//
#include <hip/hip_runtime.h>
#include <hip/hip_bf16.h>

// ---------------------------------------------------------------------------
// MI455X (gfx1250) fused cross-attention feature matching.
// All GEMMs use v_wmma_f32_16x16x32_f16 (wave32, f32 accumulate).
// Panel-hoisted WMMA operands, wave32 shuffle reductions, async LDS staging.
// ---------------------------------------------------------------------------

typedef __attribute__((ext_vector_type(16))) _Float16 v16h;
typedef __attribute__((ext_vector_type(8)))  float    v8f;

#define C_  192
#define H_  96
#define W_  256
#define B_  8
#define ROWS (B_ * H_)              // 768 (b,h) row pairs
#define ROW_HALVES (W_ * C_)        // 49152 f16 elements per row buffer

union HV { v16h v; _Float16 h[16]; };
union F8 { v8f  v; float    f[8];  };

__device__ __forceinline__ v8f wmma16(v16h a, v16h b, v8f c) {
  // 8 args: (neg_a, A, neg_b, B, c_mod, C, reuse_a, reuse_b)
  return __builtin_amdgcn_wmma_f32_16x16x32_f16(false, a, false, b, (short)0, c,
                                                false, false);
}

// Async f32 global->LDS staging (GLOBAL_LOAD_ASYNC_TO_LDS_B32, ASYNCcnt).
__device__ __forceinline__ void async_ld_f32(const float* g, float* lds) {
  unsigned off = (unsigned)(size_t)lds;   // low 32 bits = LDS byte address
  asm volatile("global_load_async_to_lds_b32 %0, %1, off"
               :: "v"(off), "v"(g) : "memory");
}
__device__ __forceinline__ void async_wait0() {
  asm volatile("s_wait_asynccnt 0" ::: "memory");
}

// A tile (16x32 f16), row-major A[m][k] at p + m*ld + k, p at tile origin.
// ISA layout: lanes 0-15: M=lane, K = {0..7, 16..23}; lanes 16-31: M=lane-16,
// K = {8..15, 24..31}.
__device__ __forceinline__ v16h load_a_rm(const _Float16* p, int ld, int lane) {
  int m  = lane & 15;
  int kb = (lane >> 4) << 3;        // 0 or 8
  const _Float16* q = p + m * ld + kb;
  HV r;
#pragma unroll
  for (int e = 0; e < 8; ++e) r.h[e]     = q[e];
#pragma unroll
  for (int e = 0; e < 8; ++e) r.h[8 + e] = q[16 + e];
  return r.v;
}

// B tile (32x16 f16) supplied from B-transposed storage: BT[n][k] row-major.
// ISA layout: N = lane&15; lanes 0-15 hold K={0..7,16..23}, lanes 16-31 hold K={8..15,24..31}.
__device__ __forceinline__ v16h load_b_bt(const _Float16* p, int ld, int lane) {
  int n  = lane & 15;
  int kb = (lane >> 4) << 3;        // 0 or 8
  const _Float16* q = p + n * ld + kb;
  HV r;
#pragma unroll
  for (int e = 0; e < 8; ++e) { r.h[e] = q[e]; r.h[8 + e] = q[16 + e]; }
  return r.v;
}

// A tile from packed-M LDS: slot[w][v] = {lo: M_r2l[w,v], hi: M_l2r[v,w]}.
// Direct orientation: A[m=w][k=v] = lo half. p at slot (w0, k0), ld in u32.
__device__ __forceinline__ v16h load_a_pack_lo(const unsigned* p, int ld, int lane) {
  int m  = lane & 15;
  int kb = (lane >> 4) << 3;
  const unsigned* q = p + m * ld + kb;
  HV r;
#pragma unroll
  for (int e = 0; e < 8; ++e) { unsigned u = q[e];      r.h[e]     = ((const _Float16*)&u)[0]; }
#pragma unroll
  for (int e = 0; e < 8; ++e) { unsigned u = q[16 + e]; r.h[8 + e] = ((const _Float16*)&u)[0]; }
  return r.v;
}

// Transposed orientation: A[m=v][k=w] = hi half of slot (w, v). p at slot (k0, v0).
__device__ __forceinline__ v16h load_a_pack_hiT(const unsigned* p, int ld, int lane) {
  int m  = lane & 15;
  int kb = (lane >> 4) << 3;
  const unsigned* q  = p + kb * ld + m;
  const unsigned* q2 = q + 16 * ld;
  HV r;
#pragma unroll
  for (int e = 0; e < 8; ++e) { unsigned u = q[e * ld];  r.h[e]     = ((const _Float16*)&u)[1]; }
#pragma unroll
  for (int e = 0; e < 8; ++e) { unsigned u = q2[e * ld]; r.h[8 + e] = ((const _Float16*)&u)[1]; }
  return r.v;
}

// ---------------------------------------------------------------------------
// Kernel 1: per (b,h) row — LayerNorm + two WMMA GEMMs (F = W_f x LN(x),
// V = W_v x x). Outputs f16: fT[w][c] (transposed) and v[c][w] (natural).
// Tile map: wt = wave/2 (B panel hoisted in regs), ot = (wave&1)*6+i.
// Called once per side (l / r). 1024 threads = 32 waves. LDS ~272 KB.
// ---------------------------------------------------------------------------
__global__ __launch_bounds__(1024) void cfm_ln_gemm_kernel(
    const float* __restrict__ x,
    const float* __restrict__ lnw_g, const float* __restrict__ lnb_g,
    const float* __restrict__ Wf, const float* __restrict__ bf,
    const float* __restrict__ Wv, const float* __restrict__ bv,
    _Float16* __restrict__ fT_base,   // [rows][W_][C_]
    _Float16* __restrict__ v_base,    // [rows][C_][W_]
    int row0) {
  extern __shared__ char smem[];
  _Float16* xT   = (_Float16*)smem;                    // [W_][C_] raw x (f16)
  _Float16* lnT  = (_Float16*)(smem + 98304);          // [W_][C_] LN(x) (f16)
  _Float16* wst  = (_Float16*)(smem + 196608);         // [C_][C_] staged weight
  float*    lnw  = (float*)(smem + 270336);            // [C_]
  float*    lnb  = lnw + C_;
  float*    bst  = lnb + C_;                           // [C_] staged bias

  const int tid  = threadIdx.x;
  const int lane = tid & 31;
  const int wv   = tid >> 5;
  const int row  = row0 + blockIdx.x;
  const int b    = row / H_;
  const int h    = row % H_;
  const float* xrow = x + ((size_t)b * C_ * H_ + h) * W_;

  // Async-stage LN params (f32, conversion-free) while x streams in.
  for (int i = tid; i < C_; i += 1024) {
    async_ld_f32(lnw_g + i, lnw + i);
    async_ld_f32(lnb_g + i, lnb + i);
  }

  // Load x row slice (coalesced in w), store transposed f16.
  for (int i = tid; i < C_ * W_; i += 1024) {
    int c = i >> 8, w = i & 255;
    xT[w * C_ + c] = (_Float16)xrow[(size_t)c * H_ * W_ + w];
  }
  async_wait0();
  __syncthreads();

  // Per-pixel LayerNorm over C: 4 lanes per pixel, wave32 shuffle combine.
  {
    int w = tid >> 2, part = tid & 3;
    const _Float16* xc = xT + w * C_;
    float s = 0.f, s2 = 0.f;
    for (int c = part; c < C_; c += 4) { float t = (float)xc[c]; s += t; s2 += t * t; }
    s  += __shfl_xor(s, 1);  s2 += __shfl_xor(s2, 1);
    s  += __shfl_xor(s, 2);  s2 += __shfl_xor(s2, 2);
    float mu  = s * (1.0f / C_);
    float var = s2 * (1.0f / C_) - mu * mu;
    float rs  = rsqrtf(var + 1e-5f);
    _Float16* lc = lnT + w * C_;
    for (int c = part; c < C_; c += 4)
      lc[c] = (_Float16)(((float)xc[c] - mu) * rs * lnw[c] + lnb[c]);
  }
  for (int i = tid; i < C_ * C_; i += 1024) wst[i] = (_Float16)Wf[i];
  for (int i = tid; i < C_;     i += 1024) bst[i] = bf[i];
  __syncthreads();

  // GEMM F = Wf[o][c] x LN[c][w]  -> fT[w][o].  12 o-tiles x 16 w-tiles.
  _Float16* fT = fT_base + (size_t)blockIdx.x * ROW_HALVES;
  {
    const int wt = wv >> 1;
    v16h breg[6];
#pragma unroll
    for (int k6 = 0; k6 < 6; ++k6)
      breg[k6] = load_b_bt(lnT + wt * 16 * C_ + k6 * 32, C_, lane);
    // Warm L2 for the second weight matrix while WMMAs run.
    for (int i = tid; i < C_ * C_; i += 1024 * 4) __builtin_prefetch(Wv + i, 0, 0);
#pragma unroll 1
    for (int i = 0; i < 6; ++i) {
      int ot = (wv & 1) * 6 + i;
      v8f acc = {};
#pragma unroll
      for (int k6 = 0; k6 < 6; ++k6)
        acc = wmma16(load_a_rm(wst + ot * 16 * C_ + k6 * 32, C_, lane),
                     breg[k6], acc);
      int n = lane & 15, mo = (lane >> 4) * 8;
      F8 d; d.v = acc;
#pragma unroll
      for (int r = 0; r < 8; ++r) {
        int o = ot * 16 + mo + r, w = wt * 16 + n;
        fT[w * C_ + o] = (_Float16)(d.f[r] + bst[o]);
      }
    }
  }
  __syncthreads();

  for (int i = tid; i < C_ * C_; i += 1024) wst[i] = (_Float16)Wv[i];
  for (int i = tid; i < C_;     i += 1024) bst[i] = bv[i];
  __syncthreads();

  // GEMM V = Wv[o][c] x x[c][w]  -> v[o][w] (natural layout).
  _Float16* vbuf = v_base + (size_t)blockIdx.x * ROW_HALVES;
  {
    const int wt = wv >> 1;
    v16h breg[6];
#pragma unroll
    for (int k6 = 0; k6 < 6; ++k6)
      breg[k6] = load_b_bt(xT + wt * 16 * C_ + k6 * 32, C_, lane);
#pragma unroll 1
    for (int i = 0; i < 6; ++i) {
      int ot = (wv & 1) * 6 + i;
      v8f acc = {};
#pragma unroll
      for (int k6 = 0; k6 < 6; ++k6)
        acc = wmma16(load_a_rm(wst + ot * 16 * C_ + k6 * 32, C_, lane),
                     breg[k6], acc);
      int n = lane & 15, mo = (lane >> 4) * 8;
      F8 d; d.v = acc;
#pragma unroll
      for (int r = 0; r < 8; ++r) {
        int o = ot * 16 + mo + r, w = wt * 16 + n;
        vbuf[o * W_ + w] = (_Float16)(d.f[r] + bst[o]);
      }
    }
  }
}

// ---------------------------------------------------------------------------
// Kernel 2: per (b,h) row — S = Fu^T Fv (WMMA), dual softmax in LDS,
// both attention-apply WMMA GEMMs fused with alpha/beta + residual.
// Softmax means are exactly 1/256 = 0.00390625 > T_THRESH, so masks == 1.
// LDS: S[256][257] f32 (padded vs bank conflicts) + stats. ~267 KB.
// ---------------------------------------------------------------------------
#define SLD 257

__global__ __launch_bounds__(1024) void cfm_attn_kernel(
    const float* __restrict__ x_l, const float* __restrict__ x_r,
    const _Float16* __restrict__ fuT_base, const _Float16* __restrict__ fvT_base,
    const _Float16* __restrict__ vl_base,  const _Float16* __restrict__ vr_base,
    const float* __restrict__ alpha, const float* __restrict__ beta,
    float* __restrict__ out_l, float* __restrict__ out_r, int row0) {
  extern __shared__ char smem[];
  float* S    = (float*)smem;                          // [256][SLD]
  float* rmax = (float*)(smem + 256 * SLD * 4);
  float* rsum = rmax + 256;
  float* cmax = rsum + 256;
  float* csum = cmax + 256;

  const int tid  = threadIdx.x;
  const int lane = tid & 31;
  const int wv   = tid >> 5;
  const int row  = row0 + blockIdx.x;
  const int b    = row / H_;
  const int h    = row % H_;

  const _Float16* fuT = fuT_base + (size_t)blockIdx.x * ROW_HALVES; // [w][c]
  const _Float16* fvT = fvT_base + (size_t)blockIdx.x * ROW_HALVES; // [v][c]
  const _Float16* vl  = vl_base  + (size_t)blockIdx.x * ROW_HALVES; // [c][w]
  const _Float16* vr  = vr_base  + (size_t)blockIdx.x * ROW_HALVES; // [c][v]

  // ---- S[w][v] = scale * sum_c Fu[c][w] Fv[c][v] : 16x16 grid of tiles.
  // Wave's A panel (its w-tile, all 6 K steps) hoisted into registers.
  const float scale = 0.07216878364870323f;            // 192^-0.5
  {
    const int wt = wv >> 1;
    v16h areg[6];
#pragma unroll
    for (int k6 = 0; k6 < 6; ++k6)
      areg[k6] = load_a_rm(fuT + wt * 16 * C_ + k6 * 32, C_, lane);
#pragma unroll 1
    for (int j = 0; j < 8; ++j) {
      int vt = (wv & 1) * 8 + j;
      v8f acc = {};
#pragma unroll
      for (int k6 = 0; k6 < 6; ++k6)
        acc = wmma16(areg[k6],
                     load_b_bt(fvT + vt * 16 * C_ + k6 * 32, C_, lane), acc);
      int n = lane & 15, mo = (lane >> 4) * 8;
      F8 d; d.v = acc;
#pragma unroll
      for (int r = 0; r < 8; ++r)
        S[(wt * 16 + mo + r) * SLD + vt * 16 + n] = d.f[r] * scale;
    }
  }
  __syncthreads();

  // ---- softmax statistics: 4 lanes per row/column, shuffle combine.
  {
    int w = tid >> 2, part = tid & 3;
    const float* Sr = S + w * SLD;
    float mx = -1e30f;
    for (int v = part; v < 256; v += 4) mx = fmaxf(mx, Sr[v]);
    mx = fmaxf(mx, __shfl_xor(mx, 1));
    mx = fmaxf(mx, __shfl_xor(mx, 2));
    float sm = 0.f;
    for (int v = part; v < 256; v += 4) sm += __expf(Sr[v] - mx);
    sm += __shfl_xor(sm, 1);
    sm += __shfl_xor(sm, 2);
    if (part == 0) { rmax[w] = mx; rsum[w] = 1.0f / sm; }
  }
  {
    int v = tid >> 2, part = tid & 3;
    float mx = -1e30f;
    for (int w = part; w < 256; w += 4) mx = fmaxf(mx, S[w * SLD + v]);
    mx = fmaxf(mx, __shfl_xor(mx, 1));
    mx = fmaxf(mx, __shfl_xor(mx, 2));
    float sm = 0.f;
    for (int w = part; w < 256; w += 4) sm += __expf(S[w * SLD + v] - mx);
    sm += __shfl_xor(sm, 1);
    sm += __shfl_xor(sm, 2);
    if (part == 0) { cmax[v] = mx; csum[v] = 1.0f / sm; }
  }
  __syncthreads();

  // ---- in-place convert: slot(w,v) := {lo: M_r2l[w,v], hi: M_l2r[v,w]} f16.
  unsigned* Su = (unsigned*)S;
  for (int i = tid; i < 65536; i += 1024) {
    int w = i >> 8, v = i & 255;
    float s = S[w * SLD + v];
    unsigned short lo = __builtin_bit_cast(unsigned short,
        (_Float16)(__expf(s - rmax[w]) * rsum[w]));
    unsigned short hi = __builtin_bit_cast(unsigned short,
        (_Float16)(__expf(s - cmax[v]) * csum[v]));
    Su[w * SLD + v] = (unsigned)lo | ((unsigned)hi << 16);
  }
  __syncthreads();

  // ---- F_r2l[c][w] = sum_v M_r2l[w,v] * Vr[c][v]; out_r = x_r + beta*F_r2l.
  // Wave: fixed w-tile (wv>>1), 6 c-tiles; A gathered once per K-half.
  {
    const int wt = wv >> 1;
    v8f acc[6] = {};
#pragma unroll 1
    for (int kh = 0; kh < 2; ++kh) {
      v16h areg[4];
#pragma unroll
      for (int k4 = 0; k4 < 4; ++k4)
        areg[k4] = load_a_pack_lo(Su + wt * 16 * SLD + kh * 128 + k4 * 32, SLD, lane);
#pragma unroll
      for (int i = 0; i < 6; ++i)
        __builtin_prefetch(vr + ((wv & 1) * 6 + i) * 16 * W_ + kh * 128 + (lane << 3), 0, 0);
#pragma unroll
      for (int i = 0; i < 6; ++i) {
        int ct = (wv & 1) * 6 + i;
#pragma unroll
        for (int k4 = 0; k4 < 4; ++k4)
          acc[i] = wmma16(areg[k4],
                          load_b_bt(vr + ct * 16 * W_ + kh * 128 + k4 * 32, W_, lane),
                          acc[i]);
      }
    }
    int n = lane & 15, mo = (lane >> 4) * 8;
#pragma unroll
    for (int i = 0; i < 6; ++i) {
      int ct = (wv & 1) * 6 + i;
      F8 d; d.v = acc[i];
#pragma unroll
      for (int r = 0; r < 8; ++r) {
        int w = wt * 16 + mo + r, c = ct * 16 + n;
        size_t gi = ((size_t)(b * C_ + c) * H_ + h) * W_ + w;
        out_r[gi] = x_r[gi] + beta[c] * d.f[r];
      }
    }
  }

  // ---- F_l2r[c][v] = sum_w M_l2r[v,w] * Vl[c][w]; out_l = x_l + alpha*F_l2r.
  {
    const int vt = wv >> 1;
    v8f acc[6] = {};
#pragma unroll 1
    for (int kh = 0; kh < 2; ++kh) {
      v16h areg[4];
#pragma unroll
      for (int k4 = 0; k4 < 4; ++k4)
        areg[k4] = load_a_pack_hiT(Su + (kh * 128 + k4 * 32) * SLD + vt * 16, SLD, lane);
#pragma unroll
      for (int i = 0; i < 6; ++i)
        __builtin_prefetch(vl + ((wv & 1) * 6 + i) * 16 * W_ + kh * 128 + (lane << 3), 0, 0);
#pragma unroll
      for (int i = 0; i < 6; ++i) {
        int ct = (wv & 1) * 6 + i;
#pragma unroll
        for (int k4 = 0; k4 < 4; ++k4)
          acc[i] = wmma16(areg[k4],
                          load_b_bt(vl + ct * 16 * W_ + kh * 128 + k4 * 32, W_, lane),
                          acc[i]);
      }
    }
    int n = lane & 15, mo = (lane >> 4) * 8;
#pragma unroll
    for (int i = 0; i < 6; ++i) {
      int ct = (wv & 1) * 6 + i;
      F8 d; d.v = acc[i];
#pragma unroll
      for (int r = 0; r < 8; ++r) {
        int v = vt * 16 + mo + r, c = ct * 16 + n;
        size_t gi = ((size_t)(b * C_ + c) * H_ + h) * W_ + v;
        out_l[gi] = x_l[gi] + alpha[c] * d.f[r];
      }
    }
  }
}

// ---------------------------------------------------------------------------
// Host launcher. Chunks rows by workspace size (deterministic in ws_size).
// Per-row ws: 4 f16 buffers of 192*256 = 384 KB.
// ---------------------------------------------------------------------------
extern "C" void kernel_launch(void* const* d_in, const int* in_sizes, int n_in,
                              void* d_out, int out_size, void* d_ws, size_t ws_size,
                              hipStream_t stream) {
  const float* x_l   = (const float*)d_in[0];
  const float* x_r   = (const float*)d_in[1];
  const float* ln1_w = (const float*)d_in[2];
  const float* ln1_b = (const float*)d_in[3];
  const float* ln2_w = (const float*)d_in[4];
  const float* ln2_b = (const float*)d_in[5];
  const float* w1    = (const float*)d_in[6];
  const float* b1    = (const float*)d_in[7];
  const float* w2    = (const float*)d_in[8];
  const float* b2    = (const float*)d_in[9];
  const float* w3    = (const float*)d_in[10];
  const float* b3    = (const float*)d_in[11];
  const float* w4    = (const float*)d_in[12];
  const float* b4    = (const float*)d_in[13];
  const float* alpha = (const float*)d_in[14];
  const float* beta  = (const float*)d_in[15];

  float* out_l = (float*)d_out;
  float* out_r = out_l + (size_t)B_ * C_ * H_ * W_;

  const size_t per_row_bytes = 4ull * ROW_HALVES * sizeof(_Float16); // 393216
  size_t chunk = ws_size / per_row_bytes;
  if (chunk < 1) chunk = 1;
  if (chunk > ROWS) chunk = ROWS;

  _Float16* fuT = (_Float16*)d_ws;
  _Float16* fvT = fuT + chunk * ROW_HALVES;
  _Float16* vl  = fvT + chunk * ROW_HALVES;
  _Float16* vr  = vl  + chunk * ROW_HALVES;

  const size_t shmem1 = 98304 + 98304 + 73728 + 3 * C_ * sizeof(float); // ~272.6 KB
  const size_t shmem2 = 256 * SLD * 4 + 4 * 256 * sizeof(float);        // ~267.3 KB

  for (int r0 = 0; r0 < ROWS; r0 += (int)chunk) {
    int g = (int)((ROWS - r0 < (int)chunk) ? (ROWS - r0) : (int)chunk);
    cfm_ln_gemm_kernel<<<g, 1024, shmem1, stream>>>(
        x_l, ln1_w, ln1_b, w1, b1, w3, b3, fuT, vl, r0);
    cfm_ln_gemm_kernel<<<g, 1024, shmem1, stream>>>(
        x_r, ln2_w, ln2_b, w2, b2, w4, b4, fvT, vr, r0);
    cfm_attn_kernel<<<g, 1024, shmem2, stream>>>(
        x_l, x_r, fuT, fvT, vl, vr, alpha, beta, out_l, out_r, r0);
  }
}
